// ProbAttention_50680614092934
// MI455X (gfx1250) — matrix operand order repaired and checked by reading, not compiled, not run on hardware
//
#include <hip/hip_runtime.h>
#include <cstdint>
#include <cstddef>

// ---------------------------------------------------------------------------
// ProbSparse attention with top_k(k==L) degenerates to exact causal attention
// (idx is a permutation; the scatter overwrites the whole cumsum).
// Flash-attention, (B=2, L=2048, H=8, D=64) fp32 in/out, bf16 WMMA compute.
//
//   * S^T = K_tile x Q^T  -> softmax stats are lane-local + one xor16 combine
//   * S^T C/D fragment == A-fragment of P for O = P x V (in-lane reshuffle)
//   * all 8 K/V LDS fragments preloaded per tile -> WMMAs run back-to-back
//   * K/V staged row-major, packed b128 LDS stores, double-buffered,
//     one barrier per tile, global loads issued ahead of the WMMA block
//   * softmax in exp2 domain (log2e folded into the Q scale, raw v_exp_f32)
//   * inline-asm v_cvt_pk_bf16_f32: one instruction per bf16 pair
//   * O rescale fully branch-skipped when the running max did not grow
// ---------------------------------------------------------------------------

typedef __attribute__((ext_vector_type(16))) __bf16   v16bf;
typedef __attribute__((ext_vector_type(8)))  float    v8f;
typedef __attribute__((ext_vector_type(4)))  float    v4f;
typedef __attribute__((ext_vector_type(4)))  uint32_t v4u;

union FragBF { v16bf v; __bf16 h[16]; uint32_t u[8]; v4u q[2]; };

constexpr int Bsz = 2;
constexpr int Lsq = 2048;
constexpr int Hh  = 8;
constexpr int Dh  = 64;

constexpr int QB  = 64;        // query rows per block
constexpr int BR  = 16;        // query rows per wave
constexpr int BC  = 32;        // key tile
constexpr int NW  = 4;         // waves per block
constexpr int SP  = Dh + 8;    // padded LDS row stride (bf16 elems, keeps 16B align)

// One v_cvt_pk_bf16_f32 per pair (the compiler otherwise emits 2 cvt + movs).
__device__ __forceinline__ uint32_t pk_bf16(float lo, float hi) {
  uint32_t u;
  asm("v_cvt_pk_bf16_f32 %0, %1, %2" : "=v"(u) : "v"(lo), "v"(hi));
  return u;
}

__device__ __forceinline__ float exp2_fast(float x) {
#if __has_builtin(__builtin_amdgcn_exp2f)
  return __builtin_amdgcn_exp2f(x);                    // raw v_exp_f32
#else
  return exp2f(x);
#endif
}

__global__ __launch_bounds__(NW * 32)
void probattn_flash_causal(const float* __restrict__ Q,
                           const float* __restrict__ K,
                           const float* __restrict__ V,
                           float* __restrict__ O)
{
  __shared__ __bf16 sK[2][BC][SP];   // K tile, row-major [key][d], double-buffered
  __shared__ __bf16 sV[2][BC][SP];   // V tile, row-major [key][d]

  const int lane = threadIdx.x & 31;
  const int w    = threadIdx.x >> 5;
  const int hf   = lane >> 4;        // lane half -> +8 row offset in C/D frags
  const int l16  = lane & 15;

  const int qtiles = Lsq / QB;
  const int qt = blockIdx.x % qtiles;
  const int bh = blockIdx.x / qtiles;
  const int h  = bh % Hh;
  const int b  = bh / Hh;

  const int rs = Hh * Dh;            // stride between sequence positions
  const size_t base = (size_t)b * Lsq * rs + (size_t)h * Dh;
  const float* Qg = Q + base;
  const float* Kg = K + base;
  const float* Vg = V + base;
  float*       Og = O + base;

  const int q0 = qt * QB + w * BR;   // this wave's first query row

  // ---- Q^T B-fragments (loop-invariant). Scale folds 1/sqrt(D) AND log2e,
  // so scores are already in log2 space and softmax uses raw v_exp_f32. ----
  const float qscale = 0.125f * 1.44269504088896340736f;
  FragBF Qb[2];
#pragma unroll
  for (int kc = 0; kc < 2; ++kc)
#pragma unroll
    for (int j = 0; j < 8; ++j) {
      const float a = qscale * Qg[(size_t)(q0 + l16) * rs + kc * 32 + 8 * hf + 2 * j + 8 * (j >> 2)];
      const float c = qscale * Qg[(size_t)(q0 + l16) * rs + kc * 32 + 8 * hf + 2 * j + 8 * (j >> 2) + 1];
      Qb[kc].u[j] = pk_bf16(a, c);
    }

  v8f Oacc[4];
#pragma unroll
  for (int nt = 0; nt < 4; ++nt) Oacc[nt] = v8f{0, 0, 0, 0, 0, 0, 0, 0};
  float mrow = -1e30f, lrow = 0.0f;  // per-lane: stats of query q0 + l16

  const int srow = threadIdx.x >> 2;           // staging: 4 threads per key row
  const int sseg = threadIdx.x & 3;
  const int nkt  = 2 * qt + 2;                 // causal tile count for block
  const int lastq = q0 + BR - 1;

  float kreg[16], vreg[16];                    // pipeline staging registers

  auto load_tile = [&](int k0) {               // global -> regs (issued early)
    const float* kp = Kg + (size_t)(k0 + srow) * rs + sseg * 16;
    const float* vp = Vg + (size_t)(k0 + srow) * rs + sseg * 16;
#pragma unroll
    for (int j = 0; j < 4; ++j) {
      v4f a = *(const v4f*)(kp + 4 * j);
      v4f c = *(const v4f*)(vp + 4 * j);
#pragma unroll
      for (int e = 0; e < 4; ++e) { kreg[4 * j + e] = a[e]; vreg[4 * j + e] = c[e]; }
    }
  };
  auto store_tile = [&](int buf) {             // regs -> LDS (bf16, packed b128)
    uint32_t pk[8], pv[8];
#pragma unroll
    for (int j = 0; j < 8; ++j) {
      pk[j] = pk_bf16(kreg[2 * j], kreg[2 * j + 1]);
      pv[j] = pk_bf16(vreg[2 * j], vreg[2 * j + 1]);
    }
    v4u* dk = (v4u*)&sK[buf][srow][sseg * 16];
    v4u* dv = (v4u*)&sV[buf][srow][sseg * 16];
    dk[0] = v4u{pk[0], pk[1], pk[2], pk[3]};
    dk[1] = v4u{pk[4], pk[5], pk[6], pk[7]};
    dv[0] = v4u{pv[0], pv[1], pv[2], pv[3]};
    dv[1] = v4u{pv[4], pv[5], pv[6], pv[7]};
  };

  load_tile(0);
  store_tile(0);
  __syncthreads();

  for (int kt = 0; kt < nkt; ++kt) {
    const int buf = kt & 1;
    const int k0  = kt * BC;
    if (kt + 1 < nkt) load_tile(k0 + BC);      // prefetch next tile into regs

    if (k0 <= lastq) {                         // skip fully-masked tail tiles
      // ---- preload ALL K/V fragments: 8 K ds_load_b128 + 64 16-bit V loads issue
      // together; distinct registers let the 8 WMMAs run without dscnt==0
      // stalls (V frags additionally hide behind the whole softmax). -------
      FragBF Ka[2][2], Vf[4];
#pragma unroll
      for (int nt = 0; nt < 2; ++nt)
#pragma unroll
        for (int kc = 0; kc < 2; ++kc) {
          const v4u* pa = (const v4u*)&sK[buf][nt * 16 + l16][kc * 32 + 8 * hf];
          const v4u* pb = (const v4u*)&sK[buf][nt * 16 + l16][kc * 32 + 8 * hf + 16];
          Ka[nt][kc].q[0] = pa[0];
          Ka[nt][kc].q[1] = pb[0];
        }
#pragma unroll
      for (int nt = 0; nt < 4; ++nt) {
#pragma unroll
        for (int s = 0; s < 16; ++s)
          Vf[nt].h[s] = sV[buf][8 * hf + s + 8 * (s >> 3)][nt * 16 + l16];
      }

      // ---- S^T = K_tile x Q^T : rows = keys, cols = queries --------------
      v8f S[2];
#pragma unroll
      for (int nt = 0; nt < 2; ++nt) {
        v8f acc = v8f{0, 0, 0, 0, 0, 0, 0, 0};
#pragma unroll
        for (int kc = 0; kc < 2; ++kc)
          acc = __builtin_amdgcn_wmma_f32_16x16x32_bf16(
              false, Ka[nt][kc].v, false, Qb[kc].v, (short)0, acc, false, false);
        S[nt] = acc;
      }

      // ---- causal mask (only diagonal tiles need it; wave-uniform) -------
      if (k0 + BC - 1 > q0) {
        const int qmy = q0 + l16;
#pragma unroll
        for (int nt = 0; nt < 2; ++nt)
#pragma unroll
          for (int r = 0; r < 8; ++r)
            if (k0 + nt * 16 + r + 8 * hf > qmy) S[nt][r] = -1e30f;
      }

      // ---- per-query online softmax (exp2 domain) ------------------------
      float mloc = S[0][0];
#pragma unroll
      for (int r = 1; r < 8; ++r) mloc = fmaxf(mloc, S[0][r]);
#pragma unroll
      for (int r = 0; r < 8; ++r) mloc = fmaxf(mloc, S[1][r]);
      mloc = fmaxf(mloc, __shfl_xor(mloc, 16, 32));

      const bool grow = __any(mloc > mrow);    // wave-uniform
      const float mold = mrow;
      const float mn = fmaxf(mrow, mloc);      // == mold when !grow
      mrow = mn;

      // S^T C/D fragment == A-fragment of P: in-lane packed converts only.
      FragBF Pa;
      float ssum = 0.0f;
#pragma unroll
      for (int j = 0; j < 4; ++j) {            // elems 0..7  -> tile nt0 keys
        const float ea = exp2_fast(S[0][2 * j]     - mn);
        const float eb = exp2_fast(S[0][2 * j + 1] - mn);
        ssum += ea + eb;
        Pa.u[j] = pk_bf16(ea, eb);
      }
#pragma unroll
      for (int j = 0; j < 4; ++j) {            // elems 8..15 -> tile nt1 keys
        const float ea = exp2_fast(S[1][2 * j]     - mn);
        const float eb = exp2_fast(S[1][2 * j + 1] - mn);
        ssum += ea + eb;
        Pa.u[4 + j] = pk_bf16(ea, eb);
      }
      ssum += __shfl_xor(ssum, 16, 32);

      // ---- rescale only when the running max actually grew ---------------
      if (grow) {
        const float alpha = exp2_fast(mold - mn);
        lrow = lrow * alpha + ssum;
        float ar[8];
#pragma unroll
        for (int r = 0; r < 8; ++r) ar[r] = __shfl(alpha, r + 8 * hf, 32);
#pragma unroll
        for (int nt = 0; nt < 4; ++nt)
#pragma unroll
          for (int r = 0; r < 8; ++r) Oacc[nt][r] *= ar[r];
      } else {
        lrow += ssum;                          // alpha == 1 exactly
      }

      // ---- O += P x V : 4 independent WMMAs, fragments already resident --
#pragma unroll
      for (int nt = 0; nt < 4; ++nt)
        Oacc[nt] = __builtin_amdgcn_wmma_f32_16x16x32_bf16(
            false, Pa.v, false, Vf[nt].v, (short)0, Oacc[nt], false, false);
    }

    if (kt + 1 < nkt) store_tile((kt + 1) & 1);  // publish next tile
    __syncthreads();                             // one barrier per tile
  }

  // ---- epilogue: normalize (broadcast 1/l into O row layout) and store ---
  const float linv = 1.0f / lrow;
  float lr[8];
#pragma unroll
  for (int r = 0; r < 8; ++r) lr[r] = __shfl(linv, r + 8 * hf, 32);
#pragma unroll
  for (int nt = 0; nt < 4; ++nt)
#pragma unroll
    for (int r = 0; r < 8; ++r)
      Og[(size_t)(q0 + r + 8 * hf) * rs + nt * 16 + l16] = Oacc[nt][r] * lr[r];
}

extern "C" void kernel_launch(void* const* d_in, const int* in_sizes, int n_in,
                              void* d_out, int out_size, void* d_ws, size_t ws_size,
                              hipStream_t stream) {
  (void)in_sizes; (void)n_in; (void)d_ws; (void)ws_size; (void)out_size;
  const float* q = (const float*)d_in[0];
  const float* k = (const float*)d_in[1];
  const float* v = (const float*)d_in[2];
  // d_in[3] (index_sample) / d_in[4] (attn_mask) cannot affect the result:
  // top_k with k == L yields a permutation and the scatter overwrites the
  // whole cumsum, reducing the op to plain causal attention.
  float* out = (float*)d_out;

  const dim3 grid(Bsz * Hh * (Lsq / QB));
  const dim3 block(NW * 32);
  probattn_flash_causal<<<grid, block, 0, stream>>>(q, k, v, out);
}
